// Wafer2SpikeResidual_83803401879948
// MI455X (gfx1250) — matrix operand order repaired and checked by reading, not compiled, not run on hardware
//
#include <hip/hip_runtime.h>
#include <hip/hip_bf16.h>

// ---------------------------------------------------------------------------
// Wafer2SpikeResidual on MI455X (gfx1250, wave32, WMMA).
// Conv+skip fused implicit GEMMs on v_wmma_f32_16x16x32_f16, tap-outer /
// channel-inner reduction over NHWC f16 spikes (no div/mod in the K loop),
// 4 accumulators per wave (16x64 strip), async global->LDS staging via
// global_load_async_to_lds_b128 (+ s_wait_asynccnt), confirmed present.
// ---------------------------------------------------------------------------

typedef _Float16 v16h __attribute__((ext_vector_type(16)));
typedef float    v8f  __attribute__((ext_vector_type(8)));
typedef int      v4i  __attribute__((ext_vector_type(4)));   // raw ext-vector (builtin type)

#define SCDECAY 0.5f
#define VDECAY  0.75f
#define VTH     0.5f

#if __has_builtin(__builtin_amdgcn_global_load_async_to_lds_b128) && \
    __has_builtin(__builtin_amdgcn_s_wait_asynccnt)
#define HAS_ASYNC_LDS 1
#else
#define HAS_ASYNC_LDS 0
#endif

// stage 16 bytes global -> LDS (async path when available)
__device__ __forceinline__ void g2l_16B(_Float16* ldst, const _Float16* gsrc) {
#if HAS_ASYNC_LDS
  __builtin_amdgcn_global_load_async_to_lds_b128(
      (__attribute__((address_space(1))) v4i*)gsrc,
      (__attribute__((address_space(3))) v4i*)ldst, 0, 0);
#else
  *reinterpret_cast<v4i*>(ldst) = *reinterpret_cast<const v4i*>(gsrc);
#endif
}

__device__ __forceinline__ void stage_wait() {
#if HAS_ASYNC_LDS
  __builtin_amdgcn_s_wait_asynccnt(0);
#endif
}

// ---------------- element-wise helpers ----------------

__global__ void zero_f32_kernel(float* __restrict__ p, long n) {
  long i = (long)blockIdx.x * blockDim.x + threadIdx.x;
  if (i < n) p[i] = 0.0f;
}

__global__ void f32_to_f16_kernel(const float* __restrict__ s,
                                  _Float16* __restrict__ d, int n) {
  int i = blockIdx.x * blockDim.x + threadIdx.x;
  if (i < n) d[i] = (_Float16)s[i];
}

// conv weights [64,64,7,7](o,c,ky,kx) x2 -> tap-major f16 [49][128][64] (tap,n,c)
__global__ void prep_conv_w_kernel(const float* __restrict__ wmain,
                                   const float* __restrict__ wskip,
                                   _Float16* __restrict__ dst) {
  int j = blockIdx.x * blockDim.x + threadIdx.x;
  if (j >= 49 * 128 * 64) return;
  int c   = j & 63;
  int n   = (j >> 6) & 127;
  int tap = j >> 13;            // 128*64 = 8192
  int o   = n & 63;
  const float* src = (n < 64) ? wmain : wskip;
  dst[j] = (_Float16)src[(o * 64 + c) * 49 + tap];
}

// stem conv (run once): x[64,1,36,36] * w[64,1,7,7] + b -> psp[64,64,30,30]
__global__ void stem_conv_kernel(const float* __restrict__ x,
                                 const float* __restrict__ w,
                                 const float* __restrict__ bias,
                                 float* __restrict__ psp) {
  int i = blockIdx.x * blockDim.x + threadIdx.x;
  if (i >= 64 * 64 * 900) return;
  int ox = i % 30;        int t1 = i / 30;
  int oy = t1 % 30;       int t2 = t1 / 30;
  int o  = t2 % 64;       int b  = t2 / 64;
  const float* xb = x + b * 36 * 36;
  const float* wo = w + o * 49;
  float acc = bias[o];
#pragma unroll
  for (int ky = 0; ky < 7; ++ky)
#pragma unroll
    for (int kx = 0; kx < 7; ++kx)
      acc += xb[(oy + ky) * 36 + (ox + kx)] * wo[ky * 7 + kx];
  psp[i] = acc;
}

// GLIF (stem): NCHW psp/state, spikes written NHWC f16 for the implicit GEMM
__global__ void glif_simple_kernel(const float* __restrict__ psp,
                                   float* __restrict__ cur,
                                   float* __restrict__ volt,
                                   float* __restrict__ spk,
                                   _Float16* __restrict__ hout,
                                   int n, int spatial) {
  int i = blockIdx.x * blockDim.x + threadIdx.x;
  if (i >= n) return;
  float c = SCDECAY * cur[i] + psp[i];
  float v = VDECAY * volt[i] * (1.0f - spk[i]) + c;
  float s = (v - VTH) > 0.0f ? 1.0f : 0.0f;
  cur[i] = c; volt[i] = v; spk[i] = s;
  int pix = i % spatial;
  int ch  = (i / spatial) & 63;
  int b   = i / (spatial * 64);
  hout[(b * spatial + pix) * 64 + ch] = (_Float16)s;   // NHWC
}

// GLIF + conv bias + skip residual; forwarded h = spike + skip (NHWC or flat)
__global__ void glif_skip_kernel(const float* __restrict__ psp,
                                 const float* __restrict__ bias,
                                 const float* __restrict__ skipv,
                                 float* __restrict__ cur,
                                 float* __restrict__ volt,
                                 float* __restrict__ spk,
                                 _Float16* __restrict__ hout,
                                 int n, int spatial, int nhwc) {
  int i = blockIdx.x * blockDim.x + threadIdx.x;
  if (i >= n) return;
  int ch = (i / spatial) & 63;
  float p = psp[i] + bias[ch];
  float c = SCDECAY * cur[i] + p;
  float v = VDECAY * volt[i] * (1.0f - spk[i]) + c;
  float s = (v - VTH) > 0.0f ? 1.0f : 0.0f;
  cur[i] = c; volt[i] = v; spk[i] = s;
  float h = s + skipv[i];
  if (nhwc) {
    int pix = i % spatial;
    int b   = i / (spatial * 64);
    hout[(b * spatial + pix) * 64 + ch] = (_Float16)h;
  } else {
    hout[i] = (_Float16)h;     // NCHW-flat (feeds FC with torch flatten order)
  }
}

// GLIF for the FC layer: bias + dropout mask on spike output (f32 kept)
__global__ void glif_fc_kernel(const float* __restrict__ psp,
                               const float* __restrict__ fcb,
                               const float* __restrict__ mask,
                               float* __restrict__ cur,
                               float* __restrict__ volt,
                               float* __restrict__ spk, int n) {
  int i = blockIdx.x * blockDim.x + threadIdx.x;
  if (i >= n) return;
  int j = i % 2304;
  float c = SCDECAY * cur[i] + psp[i] + fcb[j];
  float v = VDECAY * volt[i] * (1.0f - spk[i]) + c;
  float s = ((v - VTH) > 0.0f ? 1.0f : 0.0f) * mask[i];
  cur[i] = c; volt[i] = v; spk[i] = s;
}

// head: out[b,j] += w_t[t] * (spk[b,:] . head_w[j,:] + head_b[j])
__global__ void head_kernel(const float* __restrict__ spk,
                            const float* __restrict__ hw,
                            const float* __restrict__ hb,
                            const float* __restrict__ wt_arr, int t,
                            float* __restrict__ out) {
  int g = blockIdx.x * blockDim.x + threadIdx.x;
  if (g >= 64 * 9) return;
  int b = g / 9, j = g % 9;
  const float* s = spk + b * 2304;
  const float* w = hw + j * 2304;
  float acc = hb[j];
  for (int k = 0; k < 2304; ++k) acc += s[k] * w[k];
  out[g] += wt_arr[t] * acc;
}

// ---------------- WMMA implicit-GEMM ----------------
// Block = 128 threads = 4 waves, tile = 32M x 128N. Wave w: m-subtile (w&1),
// n-half (w>>1), 4 accumulators (16x64 strip) -> 16 WMMA per K-chunk/block.
// CONV: A is NHWC spikes, reduction ordered tap-outer / channel-inner, so
// every staged chunk is contiguous; W is tap-major [KH*KW][Ntot][IC] f16.
// Conv: n-half 0 -> out0 (main conv), n-half 1 -> out1 (skip conv), NCHW f32.
template <bool CONV>
__global__ void __launch_bounds__(128)
wmma_gemm_kernel(const _Float16* __restrict__ A, const _Float16* __restrict__ W,
                 float* __restrict__ out0, float* __restrict__ out1,
                 int Ntot, int IC, int KH, int KW,
                 int IH, int IW, int OH, int OW, int Nout) {
  __shared__ _Float16 As[32][48];    // 96B row stride: 16B-aligned sub-chunks
  __shared__ _Float16 Bs[128][48];

  const int tid  = threadIdx.x;
  const int wave = tid >> 5;
  const int lane = tid & 31;
  const int hi   = lane >> 4;
  const int lo   = lane & 15;
  const int msub  = wave & 1;
  const int nhalf = wave >> 1;
  const int spat  = OH * OW;

  // per-thread staging assignment (computed once; no index math in the loop)
  const int mA = tid >> 2;               // A row 0..31
  const int j0 = (tid & 3) * 8;          // 8-half sub-chunk
  const int rowA = blockIdx.x * 32 + mA;
  long abase;
  if (CONV) {
    int b = rowA / spat; int r2 = rowA - b * spat;
    int oy = r2 / OW;    int ox = r2 - oy * OW;
    abase = (long)((b * IH + oy * 2) * IW + ox * 2) * IC;   // NHWC pixel base
  } else {
    abase = (long)rowA * IC;
  }
  const int nglob = blockIdx.y * 128 + tid;   // B row staged by this thread

  v8f acc[4];
#pragma unroll
  for (int nt = 0; nt < 4; ++nt) acc[nt] = v8f{0.f,0.f,0.f,0.f,0.f,0.f,0.f,0.f};

  for (int ky = 0; ky < KH; ++ky) {
    for (int kx = 0; kx < KW; ++kx) {
      const long tapoff = CONV ? (long)(ky * IW + kx) * IC : 0;
      const int  tap    = ky * KW + kx;
      const _Float16* wrow = W + ((long)tap * Ntot + nglob) * IC;
      for (int c0 = 0; c0 < IC; c0 += 32) {
        // stage A (32x32) and B (128x32) tiles, all 16B-contiguous pieces
        g2l_16B(&As[mA][j0], A + abase + tapoff + c0 + j0);
        g2l_16B(&Bs[tid][0],  wrow + c0 + 0);
        g2l_16B(&Bs[tid][8],  wrow + c0 + 8);
        g2l_16B(&Bs[tid][16], wrow + c0 + 16);
        g2l_16B(&Bs[tid][24], wrow + c0 + 24);
#if !HAS_ASYNC_LDS
        if (c0 + 32 < IC) __builtin_prefetch(wrow + c0 + 32, 0, 0);
#endif
        stage_wait();
        __syncthreads();

        // fragments in contiguous k order (k = hi*16 + e, same for A and B); one A frag reused across 4 WMMAs
        v16h af;
#pragma unroll
        for (int e = 0; e < 16; ++e) {
          int ka = hi * 16 + e;
          af[e] = As[msub * 16 + lo][ka];
        }
#pragma unroll
        for (int nt = 0; nt < 4; ++nt) {
          v16h bf;
#pragma unroll
          for (int e = 0; e < 16; ++e)
            bf[e] = Bs[nhalf * 64 + nt * 16 + lo][hi * 16 + e];
          acc[nt] = __builtin_amdgcn_wmma_f32_16x16x32_f16(
              false, af, false, bf, (short)0, acc[nt], false, false);
        }
        __syncthreads();
      }
    }
  }

  // store: vgpr v -> row m0 + msub*16 + hi*8 + v; col nhalf*64 + nt*16 + lo
#pragma unroll
  for (int nt = 0; nt < 4; ++nt) {
    int col = nhalf * 64 + nt * 16 + lo;
#pragma unroll
    for (int v = 0; v < 8; ++v) {
      int row = blockIdx.x * 32 + msub * 16 + hi * 8 + v;
      if (CONV) {
        int b = row / spat; int r2 = row - b * spat;
        int oy = r2 / OW;   int ox = r2 - oy * OW;
        float* dst = nhalf ? out1 : out0;
        int o = col - nhalf * 64;
        dst[((b * 64 + o) * OH + oy) * OW + ox] = acc[nt][v];
      } else {
        out0[(long)row * Nout + blockIdx.y * 128 + col] = acc[nt][v];
      }
    }
  }
}

// ---------------- host orchestration ----------------

extern "C" void kernel_launch(void* const* d_in, const int* in_sizes, int n_in,
                              void* d_out, int out_size, void* d_ws, size_t ws_size,
                              hipStream_t stream) {
  const float* x       = (const float*)d_in[0];
  const float* stem_w  = (const float*)d_in[1];
  const float* stem_b  = (const float*)d_in[2];
  const float* b1_w    = (const float*)d_in[3];
  const float* b1_b    = (const float*)d_in[4];
  const float* b2_w    = (const float*)d_in[5];
  const float* b2_b    = (const float*)d_in[6];
  const float* skip1_w = (const float*)d_in[7];
  const float* skip2_w = (const float*)d_in[8];
  const float* fc_w    = (const float*)d_in[9];
  const float* fc_b    = (const float*)d_in[10];
  const float* head_w  = (const float*)d_in[11];
  const float* head_b  = (const float*)d_in[12];
  const float* w_t     = (const float*)d_in[13];
  const float* mask_fc = (const float*)d_in[14];
  float* out = (float*)d_out;

  const int N_STEM = 64 * 64 * 900;   // 3,686,400
  const int N_B1   = 64 * 64 * 144;   //   589,824
  const int N_B2   = 64 * 64 * 9;     //    36,864
  const int N_FC   = 64 * 2304;       //   147,456
  const int WCONV  = 49 * 128 * 64;   //   401,408 (tap-major fused conv+skip)
  const int FCW    = 2304 * 576;      // 1,327,104

  // workspace carve-out; every size is a multiple of 256B so buffers stay
  // 256B-aligned with a simple bump allocator. States first (one zero pass).
  char* ws = (char*)d_ws;
  size_t off = 0;
  auto f32buf = [&](size_t n) { float* p = (float*)(ws + off); off += n * sizeof(float); return p; };
  auto f16buf = [&](size_t n) { _Float16* p = (_Float16*)(ws + off); off += n * sizeof(_Float16); return p; };

  float* states = (float*)(ws + off);
  float *stem_cur = f32buf(N_STEM), *stem_volt = f32buf(N_STEM), *stem_spk = f32buf(N_STEM);
  float *b1_cur = f32buf(N_B1),  *b1_volt = f32buf(N_B1),  *b1_spk = f32buf(N_B1);
  float *b2_cur = f32buf(N_B2),  *b2_volt = f32buf(N_B2),  *b2_spk = f32buf(N_B2);
  float *fc_cur = f32buf(N_FC),  *fc_volt = f32buf(N_FC),  *fc_spk = f32buf(N_FC);
  const long nstate = 3L * (N_STEM + N_B1 + N_B2 + N_FC);

  float* stem_psp = f32buf(N_STEM);
  float* b1_psp   = f32buf(N_B1);
  float* skip1_o  = f32buf(N_B1);
  float* b2_psp   = f32buf(N_B2);
  float* skip2_o  = f32buf(N_B2);
  float* fc_psp   = f32buf(N_FC);
  _Float16* h0_16 = f16buf(N_STEM);          // NHWC [64,30,30,64]
  _Float16* h1_16 = f16buf(N_B1);            // NHWC [64,12,12,64]
  _Float16* h2_16 = f16buf(N_B2);            // NCHW-flat [64,576] (FC order)
  _Float16* wc1   = f16buf(WCONV);           // [49][128][64]
  _Float16* wc2   = f16buf(WCONV);
  _Float16* fcw16 = f16buf(FCW);             // [2304][576]
  (void)ws_size; (void)in_sizes; (void)n_in; (void)out_size;

  const int BS = 256;
  auto g = [](long n, int bs) { return dim3((unsigned)((n + bs - 1) / bs)); };

  // deterministic re-init every call
  zero_f32_kernel<<<g(nstate, BS), BS, 0, stream>>>(states, nstate);
  zero_f32_kernel<<<g(576, BS), BS, 0, stream>>>(out, 576);

  // weight prep
  f32_to_f16_kernel<<<g(FCW, BS), BS, 0, stream>>>(fc_w, fcw16, FCW);
  prep_conv_w_kernel<<<g(WCONV, BS), BS, 0, stream>>>(b1_w, skip1_w, wc1);
  prep_conv_w_kernel<<<g(WCONV, BS), BS, 0, stream>>>(b2_w, skip2_w, wc2);

  // stem conv once (constant input encoding)
  stem_conv_kernel<<<g(N_STEM, BS), BS, 0, stream>>>(x, stem_w, stem_b, stem_psp);

  for (int t = 0; t < 16; ++t) {
    glif_simple_kernel<<<g(N_STEM, BS), BS, 0, stream>>>(
        stem_psp, stem_cur, stem_volt, stem_spk, h0_16, N_STEM, 900);

    // b1 + skip1: M=9216, N=128, K=3136 (49 taps x 64 ch)
    wmma_gemm_kernel<true><<<dim3(9216 / 32, 1), 128, 0, stream>>>(
        h0_16, wc1, b1_psp, skip1_o, 128, 64, 7, 7, 30, 30, 12, 12, 0);
    glif_skip_kernel<<<g(N_B1, BS), BS, 0, stream>>>(
        b1_psp, b1_b, skip1_o, b1_cur, b1_volt, b1_spk, h1_16, N_B1, 144, 1);

    // b2 + skip2: M=576, N=128, K=3136
    wmma_gemm_kernel<true><<<dim3(576 / 32, 1), 128, 0, stream>>>(
        h1_16, wc2, b2_psp, skip2_o, 128, 64, 7, 7, 12, 12, 3, 3, 0);
    glif_skip_kernel<<<g(N_B2, BS), BS, 0, stream>>>(
        b2_psp, b2_b, skip2_o, b2_cur, b2_volt, b2_spk, h2_16, N_B2, 9, 0);

    // FC: M=64, N=2304, K=576
    wmma_gemm_kernel<false><<<dim3(64 / 32, 2304 / 128), 128, 0, stream>>>(
        h2_16, fcw16, fc_psp, nullptr, 2304, 576, 1, 1, 0, 0, 1, 1, 2304);
    glif_fc_kernel<<<g(N_FC, BS), BS, 0, stream>>>(
        fc_psp, fc_b, mask_fc, fc_cur, fc_volt, fc_spk, N_FC);

    head_kernel<<<g(576, BS), BS, 0, stream>>>(fc_spk, head_w, head_b, w_t, t, out);
  }
}
